// JordanBeliefPropagation_43224550867284
// MI455X (gfx1250) — matrix operand order repaired and checked by reading, not compiled, not run on hardware
//
#include <hip/hip_runtime.h>
#include <math.h>

typedef __attribute__((ext_vector_type(16))) _Float16 v16h;
typedef __attribute__((ext_vector_type(8)))  _Float16 v8h;
typedef __attribute__((ext_vector_type(8)))  float    v8f;
typedef __attribute__((ext_vector_type(4)))  float    v4f;
typedef __attribute__((ext_vector_type(2)))  float    v2f;

#define NA 16   // agents
#define DD 27   // state dim
#define DP 28   // padded state dim (messages rows)
#define D2 54   // 2*DD

__device__ __forceinline__ v4f splat4(float x) { v4f v = {x, x, x, x}; return v; }

// out[0..27] = a[0..26] @ Jp  (Jp is [32][32] in LDS, rows/cols >=27 are 0)
__device__ __forceinline__ void matvecJ32(const float* a, float* out, const float* Jp) {
#pragma unroll
  for (int dv = 0; dv < 7; ++dv) {
    v4f s = {0.f, 0.f, 0.f, 0.f};
#pragma unroll
    for (int c = 0; c < DD; ++c)
      s += splat4(a[c]) * (*reinterpret_cast<const v4f*>(Jp + c * 32 + dv * 4));
    *reinterpret_cast<v4f*>(out + dv * 4) = s;
  }
}

// ---------------- f32 WMMA batched matvec: O[256][32] = A[256][K=32] @ Jp[32][32]
// A rows have stride ASTRIDE floats (28 for messages, 32 for AGG).
// f32 A frag (16x4): lanes 0-15: M=lane, K={kp,kp+1}; lanes 16-31: M=lane-16, K={kp+2,kp+3}
// f32 B frag (4x16): lanes 0-15: N=lane, K={kp,kp+1}; lanes 16-31: N=lane-16, K={kp+2,kp+3}
template <int ASTRIDE>
__device__ __forceinline__ void gemm_matvec_f32(const float* Ab, const float* Jp,
                                                float* Ob, int tid) {
  const int wave = tid >> 5;
  const int lane = tid & 31;
  const int lrow = lane & 15;
  const int hi2 = (lane >> 4) << 1;
  const int hi8 = (lane >> 4) << 3;
  const v8f zero = {0.f, 0.f, 0.f, 0.f, 0.f, 0.f, 0.f, 0.f};
#pragma unroll
  for (int mi = 0; mi < 2; ++mi) {
    const int mt = wave * 2 + mi;
    v8f acc0 = zero, acc1 = zero;
#pragma unroll
    for (int ks = 0; ks < 8; ++ks) {
      const int kp = ks * 4 + hi2;
      v2f a = *reinterpret_cast<const v2f*>(Ab + (mt * 16 + lrow) * ASTRIDE + kp);
      v2f b0 = {Jp[kp * 32 + lrow], Jp[(kp + 1) * 32 + lrow]};
      v2f b1 = {Jp[kp * 32 + 16 + lrow], Jp[(kp + 1) * 32 + 16 + lrow]};
      acc0 = __builtin_amdgcn_wmma_f32_16x16x4_f32(false, a, false, b0, (short)0,
                                                   acc0, false, false);
      acc1 = __builtin_amdgcn_wmma_f32_16x16x4_f32(false, a, false, b1, (short)0,
                                                   acc1, false, false);
    }
#pragma unroll
    for (int r = 0; r < 8; ++r) {
      int m = mt * 16 + r + hi8;
      Ob[m * 32 + lrow] = acc0[r];
      Ob[m * 32 + 16 + lrow] = acc1[r];
    }
  }
}

// ---------------- f16 WMMA fragments for the MLP GEMMs ----------------
__device__ __forceinline__ v16h load_fragA(const _Float16* base, int mbase, int ks, int lane) {
  int m = mbase + (lane & 15);
  int kofs = ks + ((lane >> 4) << 3);
  const _Float16* p = base + m * 64 + kofs;
  v8h lo = *reinterpret_cast<const v8h*>(p);
  v8h hi = *reinterpret_cast<const v8h*>(p + 16);
  v16h r;
#pragma unroll
  for (int q = 0; q < 8; ++q) { r[q] = lo[q]; r[q + 8] = hi[q]; }
  return r;
}

__device__ __forceinline__ v16h load_fragB(const _Float16* base, int nbase, int ks, int lane) {
  int n = nbase + (lane & 15);
  int kofs = ks + ((lane >> 4) << 3);
  const _Float16* p = base + n * 64 + kofs;
  v8h lo = *reinterpret_cast<const v8h*>(p);
  v8h hi = *reinterpret_cast<const v8h*>(p + 16);
  v16h r;
#pragma unroll
  for (int q = 0; q < 8; ++q) { r[q] = lo[q]; r[q + 8] = hi[q]; }
  return r;
}

template <int NTILES>
__device__ __forceinline__ void do_gemm(const _Float16* Ab, const _Float16* Bmat,
                                        _Float16* Hb, int tid) {
  const int wave = tid >> 5;
  const int lane = tid & 31;
  const int lrow = lane & 15;
  const int hi8 = (lane >> 4) << 3;
  const v8f zero = {0.f, 0.f, 0.f, 0.f, 0.f, 0.f, 0.f, 0.f};
#pragma unroll
  for (int mi = 0; mi < 2; ++mi) {
    const int mt = wave * 2 + mi;
    v8f acc[NTILES];
#pragma unroll
    for (int nt = 0; nt < NTILES; ++nt) acc[nt] = zero;
#pragma unroll
    for (int ks = 0; ks < 2; ++ks) {
      v16h a = load_fragA(Ab, mt * 16, ks * 32, lane);
#pragma unroll
      for (int nt = 0; nt < NTILES; ++nt) {
        v16h bf = load_fragB(Bmat, nt * 16, ks * 32, lane);
        acc[nt] = __builtin_amdgcn_wmma_f32_16x16x32_f16(
            false, a, false, bf, (short)0, acc[nt], false, false);
      }
    }
#pragma unroll
    for (int nt = 0; nt < NTILES; ++nt) {
#pragma unroll
      for (int r = 0; r < 8; ++r) {
        int m = mt * 16 + r + hi8;
        Hb[m * 64 + nt * 16 + lrow] = (_Float16)acc[nt][r];
      }
    }
  }
}

__global__ __launch_bounds__(256, 1) void jbp_kernel(
    const float* __restrict__ g_bel0, const float* __restrict__ g_S,
    const float* __restrict__ g_potl, const float* __restrict__ g_w1,
    const float* __restrict__ g_b1, const float* __restrict__ g_lnw,
    const float* __restrict__ g_lnb, const float* __restrict__ g_w2,
    const float* __restrict__ g_b2, const float* __restrict__ g_aggw,
    const float* __restrict__ g_aggb, float* __restrict__ g_out) {
  __shared__ float s_msg[NA * NA * DP];                         // 28.0 KB  m[k][r][d]
  __shared__ __align__(16) unsigned char s_Abuf[256 * 128];     // 32 KB AGG f32 / MLP-A f16
  __shared__ __align__(16) unsigned char s_scr[256 * 128];      // 32 KB JM f32 / H f16
  __shared__ float s_aggj[256 * 32];                            // 32 KB AGG@J result
  __shared__ _Float16 s_B1[64 * 64];                            // 8 KB  w1 as [n][k]
  __shared__ _Float16 s_B2[32 * 64];                            // 4 KB  w2 as [n][k]
  __shared__ float s_Jp[32 * 32];                               // 4 KB  J padded
  __shared__ float s_aggw[DD][56];
  __shared__ float s_bel[NA][DP];
  __shared__ float s_Jb[NA][DP];
  __shared__ float s_inc[NA][DP];
  __shared__ float s_upd[NA][DP];
  __shared__ float s_pot[NA][NA];
  __shared__ float s_b1v[64], s_lnw[64], s_lnb[64], s_b2v[32], s_aggb[32];

  float* AGGf = reinterpret_cast<float*>(s_Abuf);       // [256][32]
  _Float16* A16 = reinterpret_cast<_Float16*>(s_Abuf);  // [256][64]
  float* JMf = reinterpret_cast<float*>(s_scr);         // [256][32]  (row = k*16+r)
  _Float16* H16 = reinterpret_cast<_Float16*>(s_scr);   // [256][64]

  const int tid = threadIdx.x;
  const int b = blockIdx.x;
  const int pi = tid >> 4;
  const int pj = tid & 15;
  // first k not in {pi, pj} (ascending fold order)
  const int fv = (pi != 0 && pj != 0) ? 0 : ((pi != 1 && pj != 1) ? 1 : 2);

  // ---------------- parameter / state load ----------------
  for (int idx = tid; idx < 32 * 32; idx += 256) {
    int r = idx >> 5, c = idx & 31;
    s_Jp[idx] = (r < DD && c < DD) ? 0.5f * (g_S[r * DD + c] + g_S[c * DD + r]) : 0.f;
  }
  for (int idx = tid; idx < NA * NA; idx += 256)
    s_pot[idx >> 4][idx & 15] = 1.f / (1.f + expf(-g_potl[idx]));
  for (int idx = tid; idx < 64 * 64; idx += 256) {
    int n = idx >> 6, k = idx & 63;
    s_B1[idx] = (n < D2 && k < D2) ? (_Float16)g_w1[n * D2 + k] : (_Float16)0.f;
  }
  for (int idx = tid; idx < 32 * 64; idx += 256) {
    int n = idx >> 6, k = idx & 63;
    s_B2[idx] = (n < DD && k < D2) ? (_Float16)g_w2[n * D2 + k] : (_Float16)0.f;
  }
  for (int idx = tid; idx < DD * D2; idx += 256)
    s_aggw[idx / D2][idx % D2] = g_aggw[idx];
  if (tid < 64) {
    s_b1v[tid] = (tid < D2) ? g_b1[tid] : 0.f;
    s_lnw[tid] = (tid < D2) ? g_lnw[tid] : 0.f;
    s_lnb[tid] = (tid < D2) ? g_lnb[tid] : 0.f;
    if (tid < 32) {
      s_b2v[tid] = (tid < DD) ? g_b2[tid] : 0.f;
      s_aggb[tid] = (tid < DD) ? g_aggb[tid] : 0.f;
    }
  }
  for (int idx = tid; idx < NA * DD; idx += 256)
    s_bel[idx / DD][idx % DD] = g_bel0[(size_t)b * NA * DD + idx];
  __syncthreads();

  // messages init: m_{i->j} = beliefs[i]; pad col zeroed (keeps WMMA pads exact)
  {
    float* mrow = s_msg + tid * DP;
#pragma unroll
    for (int d = 0; d < DD; ++d) mrow[d] = s_bel[pi][d];
    mrow[DD] = 0.f;
  }
  __syncthreads();

  for (int it = 0; it < 3; ++it) {
    // ---- JM[k*16+r] = m[k][r] @ J  (batched on the matrix pipe, fp32) ----
    gemm_matvec_f32<DP>(s_msg, s_Jp, JMf, tid);
    // ---- Jb[r] = beliefs[r] @ J ----
    if (tid < NA) {
      float bl[DD];
#pragma unroll
      for (int d = 0; d < DD; ++d) bl[d] = s_bel[tid][d];
      float o[DP];
      matvecJ32(bl, o, s_Jp);
#pragma unroll
      for (int d = 0; d < DD; ++d) s_Jb[tid][d] = o[d];
    }
    __syncthreads();

    // ---- order-sensitive fold, matvec on WMMA-f32, select/update elementwise ----
    for (int k = 0; k < NA; ++k) {
      if (k >= 1) gemm_matvec_f32<32>(AGGf, s_Jp, s_aggj, tid);  // AGGJ = AGG @ J
      __syncthreads();
      if (k != pi && k != pj) {
        float* arow = AGGf + tid * 32;
        const float* mk = s_msg + (k * NA + pi) * DP;
        if (k == fv) {
#pragma unroll
          for (int d = 0; d < DD; ++d) arow[d] = mk[d];
#pragma unroll
          for (int d = DD; d < 32; ++d) arow[d] = 0.f;  // exact-zero pads
        } else {
          const float* jm = JMf + (k * NA + pi) * 32;
          const float* aj = s_aggj + tid * 32;
#pragma unroll
          for (int d = 0; d < DD; ++d)
            arow[d] = 0.5f * (arow[d] * jm[d] + mk[d] * aj[d]);
        }
      }
      __syncthreads();
    }

    // ---- combined = JP(beliefs[i], agg); stage msg_in row as f16 (in place) ----
    gemm_matvec_f32<32>(AGGf, s_Jp, s_aggj, tid);
    __syncthreads();
    {
      float aggv[DD];
      const float* arow = AGGf + tid * 32;
#pragma unroll
      for (int d = 0; d < DD; ++d) aggv[d] = arow[d];
      const float* aj = s_aggj + tid * 32;
      _Float16* a16 = A16 + tid * 64;
#pragma unroll
      for (int d = 0; d < DD; ++d) {
        float comb = 0.5f * (s_bel[pi][d] * aj[d] + aggv[d] * s_Jb[pi][d]);
        a16[d] = (_Float16)comb;
        a16[DD + d] = (_Float16)s_bel[pj][d];
      }
#pragma unroll
      for (int d = 2 * DD; d < 64; ++d) a16[d] = (_Float16)0.f;
    }
    __syncthreads();

    // ---- GEMM1: h = msg_in @ w1^T ----
    do_gemm<4>(A16, s_B1, H16, tid);
    __syncthreads();

    // ---- +b1, LayerNorm, exact GELU; rewrite as GEMM2 A operand ----
    {
      float h[D2];
      const _Float16* hr = H16 + tid * 64;
      float mu = 0.f;
#pragma unroll
      for (int n = 0; n < D2; ++n) {
        h[n] = (float)hr[n] + s_b1v[n];
        mu += h[n];
      }
      mu *= (1.f / (float)D2);
      float var = 0.f;
#pragma unroll
      for (int n = 0; n < D2; ++n) {
        float dl = h[n] - mu;
        var += dl * dl;
      }
      var *= (1.f / (float)D2);
      float rstd = rsqrtf(var + 1e-5f);
      _Float16* a16 = A16 + tid * 64;
#pragma unroll
      for (int n = 0; n < D2; ++n) {
        float x = (h[n] - mu) * rstd * s_lnw[n] + s_lnb[n];
        float g = 0.5f * x * (1.f + erff(x * 0.70710678118654752f));
        a16[n] = (_Float16)g;
      }
#pragma unroll
      for (int n = D2; n < 64; ++n) a16[n] = (_Float16)0.f;
    }
    __syncthreads();

    // ---- GEMM2: h @ w2^T ----
    do_gemm<2>(A16, s_B2, H16, tid);
    __syncthreads();

    // ---- new messages: (out + b2) * pot[i][j], diagonal zeroed ----
    {
      float p = s_pot[pi][pj];
      const _Float16* hr = H16 + tid * 64;
      float* mrow = s_msg + tid * DP;
#pragma unroll
      for (int d = 0; d < DD; ++d) {
        float v = ((float)hr[d] + s_b2v[d]) * p;
        mrow[d] = (pi == pj) ? 0.f : v;
      }
    }
    __syncthreads();

    // ---- incoming[r] = sum_k m[k][r] ----
    for (int idx = tid; idx < NA * DD; idx += 256) {
      int r = idx / DD, d = idx % DD;
      float s = 0.f;
#pragma unroll
      for (int k = 0; k < NA; ++k) s += s_msg[(k * NA + r) * DP + d];
      s_inc[r][d] = s;
    }
    __syncthreads();

    // ---- updated = JP(beliefs, incoming) ----
    if (tid < NA) {
      float inc[DD];
#pragma unroll
      for (int d = 0; d < DD; ++d) inc[d] = s_inc[tid][d];
      float Ji[DP];
      matvecJ32(inc, Ji, s_Jp);
#pragma unroll
      for (int d = 0; d < DD; ++d)
        s_upd[tid][d] = 0.5f * (s_bel[tid][d] * Ji[d] + inc[d] * s_Jb[tid][d]);
    }
    __syncthreads();

    // ---- beliefs = normalize([beliefs ++ updated] @ agg_w^T + agg_b) ----
    if (tid < NA) {
      float nb[DD];
      float ss = 0.f;
#pragma unroll
      for (int o = 0; o < DD; ++o) {
        float s = s_aggb[o];
        for (int c = 0; c < DD; ++c)
          s += s_bel[tid][c] * s_aggw[o][c] + s_upd[tid][c] * s_aggw[o][DD + c];
        nb[o] = s;
        ss += s * s;
      }
      float inv = 1.f / fmaxf(sqrtf(ss), 1e-8f);
#pragma unroll
      for (int o = 0; o < DD; ++o) s_bel[tid][o] = nb[o] * inv;
    }
    __syncthreads();
  }

  for (int idx = tid; idx < NA * DD; idx += 256)
    g_out[(size_t)b * NA * DD + idx] = s_bel[idx / DD][idx % DD];
}

extern "C" void kernel_launch(void* const* d_in, const int* in_sizes, int n_in,
                              void* d_out, int out_size, void* d_ws, size_t ws_size,
                              hipStream_t stream) {
  (void)n_in; (void)out_size; (void)d_ws; (void)ws_size;
  const float* bel0 = (const float*)d_in[0];
  const float* S    = (const float*)d_in[1];
  const float* potl = (const float*)d_in[2];
  const float* w1   = (const float*)d_in[3];
  const float* b1   = (const float*)d_in[4];
  const float* lnw  = (const float*)d_in[5];
  const float* lnb  = (const float*)d_in[6];
  const float* w2   = (const float*)d_in[7];
  const float* b2   = (const float*)d_in[8];
  const float* aggw = (const float*)d_in[9];
  const float* aggb = (const float*)d_in[10];
  float* out = (float*)d_out;

  int Bb = in_sizes[0] / (NA * DD);
  jbp_kernel<<<Bb, 256, 0, stream>>>(bel0, S, potl, w1, b1, lnw, lnb, w2, b2,
                                     aggw, aggb, out);
}
